// LSTM_41918880809220
// MI455X (gfx1250) — matrix operand order repaired and checked by reading, not compiled, not run on hardware
//
#include <hip/hip_runtime.h>

typedef _Float16 v8h  __attribute__((ext_vector_type(8)));
typedef _Float16 v16h __attribute__((ext_vector_type(16)));
typedef float    v8f  __attribute__((ext_vector_type(8)));
typedef int      i4v  __attribute__((vector_size(16)));

#define HID 128
#define SEQT 28

#if __has_builtin(__builtin_amdgcn_global_load_async_to_lds_b128)
#define USE_ASYNC_LDS 1
#endif

__device__ __forceinline__ float sigm_(float x) {
    return 1.0f / (1.0f + __expf(-x));
}
__device__ __forceinline__ float tanh_(float x) {
    x = fminf(fmaxf(x, -15.0f), 15.0f);
    float e = __expf(-2.0f * x);
    return (1.0f - e) / (1.0f + e);
}

// 16-byte global -> LDS copy: async DMA (ASYNCcnt) when available.
__device__ __forceinline__ void copy16_g2l(const _Float16* g, _Float16* l) {
#ifdef USE_ASYNC_LDS
    __builtin_amdgcn_global_load_async_to_lds_b128(
        (__attribute__((address_space(1))) i4v*)g,
        (__attribute__((address_space(3))) i4v*)l, 0, 0);
#else
    *(uint4*)l = *(const uint4*)g;
#endif
}
__device__ __forceinline__ void wait_async0() {
#ifdef USE_ASYNC_LDS
#if __has_builtin(__builtin_amdgcn_s_wait_asynccnt)
    __builtin_amdgcn_s_wait_asynccnt(0);
#else
    asm volatile("s_wait_asynccnt 0" ::: "memory");
#endif
#endif
}

// ---------------------------------------------------------------------------
// Prep: weights f32->f16 (pad layer0 K 28->32), fold biases, and convert the
// input images to padded f16 [N][28][32] so per-step staging is a pure DMA.
// ---------------------------------------------------------------------------
__global__ void prep_kernel(const float* __restrict__ x, int nrows /* N*28 */,
                            const float* __restrict__ wih0, const float* __restrict__ whh0,
                            const float* __restrict__ bih0, const float* __restrict__ bhh0,
                            const float* __restrict__ wih1, const float* __restrict__ whh1,
                            const float* __restrict__ bih1, const float* __restrict__ bhh1,
                            _Float16* __restrict__ x16,
                            _Float16* __restrict__ wih0f, _Float16* __restrict__ whh0f,
                            _Float16* __restrict__ wih1f, _Float16* __restrict__ whh1f,
                            float* __restrict__ bias0, float* __restrict__ bias1)
{
    int tid = blockIdx.x * blockDim.x + threadIdx.x;
    int nth = gridDim.x * blockDim.x;
    for (size_t i = tid; i < (size_t)nrows * 32; i += nth) {
        size_t r = i >> 5; int c = (int)(i & 31);
        x16[i] = (_Float16)((c < 28) ? x[r * 28 + c] : 0.0f);
    }
    for (int i = tid; i < 512 * 32; i += nth) {
        int n = i >> 5, k = i & 31;
        wih0f[i] = (_Float16)((k < 28) ? wih0[n * 28 + k] : 0.0f);
    }
    for (int i = tid; i < 512 * HID; i += nth) whh0f[i] = (_Float16)whh0[i];
    for (int i = tid; i < 512 * HID; i += nth) wih1f[i] = (_Float16)wih1[i];
    for (int i = tid; i < 512 * HID; i += nth) whh1f[i] = (_Float16)whh1[i];
    for (int i = tid; i < 512; i += nth) bias0[i] = bih0[i] + bhh0[i];
    for (int i = tid; i < 512; i += nth) bias1[i] = bih1[i] + bhh1[i];
}

// ---------------------------------------------------------------------------
// One LSTM layer fused over T=28 steps. Block = 256 threads = 8 waves,
// 64 batch rows per block. Wave w owns hidden cols [w*16, w*16+16) and all 4
// of their gate tiles, so c/h updates stay entirely in registers. Input tiles
// are double-buffered in LDS and prefetched with async global->LDS DMA.
//
// WMMA fragment layouts (CDNA5 ISA 7.12.2):
//   A (16x32 f16): lane<16 -> K j: 0..7,16..23 ; lane>=16 -> 8..15,24..31
//   B (32x16 f16): lane holds col n=lane&15; code loads A and B as K=(lane>>4)*16+j
//   C/D (16x16 f32): vgpr r -> m = r + 8*(lane>>4), n = lane&15
// ---------------------------------------------------------------------------
template<int LAYER>
__global__ void __launch_bounds__(256) lstm_layer(
    const _Float16* __restrict__ hin,   // f16 [N][28][KX] (x16 or hs1)
    const _Float16* __restrict__ Wx,    // [512][KX] f16
    const _Float16* __restrict__ Wh,    // [512][128] f16
    const float*    __restrict__ bias,  // [512]
    _Float16*       __restrict__ hs_out,// LAYER==0: [N][28][128] f16
    const float*    __restrict__ wout,  // LAYER==1: [10][128]
    const float*    __restrict__ bout,  // LAYER==1: [10]
    float*          __restrict__ out)   // LAYER==1: [N][10]
{
    constexpr int KX  = (LAYER == 0) ? 32 : HID;
    constexpr int KXT = KX / 32;
    constexpr int NCH = KX / 8;                 // 16B chunks per row

    __shared__ _Float16 sX[2][64 * KX];
    __shared__ _Float16 sH[64 * HID];

    const int tid  = threadIdx.x;
    const int wave = tid >> 5;
    const int lane = tid & 31;
    const int lrow = lane & 15;
    const int lhi  = lane >> 4;
    const int row0 = blockIdx.x * 64;

    // --- persistent B fragments (whole time loop) ---
    v16h bx[4][KXT];
    v16h bh[4][4];
    #pragma unroll
    for (int g = 0; g < 4; ++g) {
        const int n = g * HID + wave * 16 + lrow;     // gate output column
        #pragma unroll
        for (int kt = 0; kt < KXT; ++kt)
            bx[g][kt] = *(const v16h*)(Wx + (size_t)n * KX + kt * 32 + lhi * 16);
        #pragma unroll
        for (int kt = 0; kt < 4; ++kt)
            bh[g][kt] = *(const v16h*)(Wh + (size_t)n * HID + kt * 32 + lhi * 16);
    }

    float bg[4];
    #pragma unroll
    for (int g = 0; g < 4; ++g)
        bg[g] = bias[g * HID + wave * 16 + lrow];

    float cst[4][8];
    #pragma unroll
    for (int mt = 0; mt < 4; ++mt)
        #pragma unroll
        for (int r = 0; r < 8; ++r) cst[mt][r] = 0.0f;

    for (int i = tid; i < 64 * HID; i += 256) sH[i] = (_Float16)0.0f;

    // stage x_0 into buffer 0
    for (int i = tid; i < 64 * NCH; i += 256) {
        int r = i / NCH, q = i % NCH;
        copy16_g2l(hin + ((size_t)(row0 + r) * SEQT + 0) * KX + q * 8,
                   &sX[0][r * KX + q * 8]);
    }
    wait_async0();   // own chunks landed; cross-wave visibility via barrier below

    #pragma unroll 1
    for (int t = 0; t < SEQT; ++t) {
        const int cur = t & 1, nxt = cur ^ 1;

        // ---- issue async prefetch of x_{t+1} into the other buffer ----
        if (t + 1 < SEQT) {
            for (int i = tid; i < 64 * NCH; i += 256) {
                int r = i / NCH, q = i % NCH;
                copy16_g2l(hin + ((size_t)(row0 + r) * SEQT + (t + 1)) * KX + q * 8,
                           &sX[nxt][r * KX + q * 8]);
            }
        }
        __syncthreads();   // sX[cur] staged everywhere, sH holds h_{t-1}

        // ---- gates = bias + x_t*Wx^T + h*Wh^T ----
        v8f acc[4][4];
        #pragma unroll
        for (int g = 0; g < 4; ++g)
            #pragma unroll
            for (int mt = 0; mt < 4; ++mt)
                #pragma unroll
                for (int r = 0; r < 8; ++r) acc[g][mt][r] = bg[g];

        #pragma unroll
        for (int kt = 0; kt < KXT; ++kt) {
            #pragma unroll
            for (int mt = 0; mt < 4; ++mt) {
                const _Float16* arow = &sX[cur][(mt * 16 + lrow) * KX + kt * 32];
                v16h a = *(const v16h*)(arow + lhi * 16);
                #pragma unroll
                for (int g = 0; g < 4; ++g)
                    acc[g][mt] = __builtin_amdgcn_wmma_f32_16x16x32_f16(
                        false, a, false, bx[g][kt], (short)0, acc[g][mt], false, false);
            }
        }
        #pragma unroll
        for (int kt = 0; kt < 4; ++kt) {
            #pragma unroll
            for (int mt = 0; mt < 4; ++mt) {
                const _Float16* arow = sH + (mt * 16 + lrow) * HID + kt * 32;
                v16h a = *(const v16h*)(arow + lhi * 16);
                #pragma unroll
                for (int g = 0; g < 4; ++g)
                    acc[g][mt] = __builtin_amdgcn_wmma_f32_16x16x32_f16(
                        false, a, false, bh[g][kt], (short)0, acc[g][mt], false, false);
            }
        }

        // ---- pointwise LSTM cell update (all in registers) ----
        float hv[4][8];
        #pragma unroll
        for (int mt = 0; mt < 4; ++mt) {
            #pragma unroll
            for (int r = 0; r < 8; ++r) {
                float iv = sigm_(acc[0][mt][r]);
                float fv = sigm_(acc[1][mt][r]);
                float gv = tanh_(acc[2][mt][r]);
                float ov = sigm_(acc[3][mt][r]);
                float c  = fv * cst[mt][r] + iv * gv;
                cst[mt][r] = c;
                hv[mt][r]  = ov * tanh_(c);
            }
        }

        wait_async0();     // this wave's prefetch chunks are in LDS
        __syncthreads();   // all prefetches landed; all reads of sH done

        #pragma unroll
        for (int mt = 0; mt < 4; ++mt)
            #pragma unroll
            for (int r = 0; r < 8; ++r) {
                int rr = mt * 16 + lhi * 8 + r;
                _Float16 hh = (_Float16)hv[mt][r];
                sH[rr * HID + wave * 16 + lrow] = hh;
                if (LAYER == 0)
                    hs_out[((size_t)(row0 + rr) * SEQT + t) * HID + wave * 16 + lrow] = hh;
            }
    }

    if (LAYER == 1) {
        __syncthreads();   // sH = h_{T-1}
        for (int i = tid; i < 64 * 10; i += 256) {
            int r = i / 10, cls = i % 10;
            float s = bout[cls];
            const _Float16* hrow = sH + r * HID;
            #pragma unroll 8
            for (int k = 0; k < HID; ++k)
                s += (float)hrow[k] * wout[cls * HID + k];
            out[(size_t)(row0 + r) * 10 + cls] = s;
        }
    }
}

// ---------------------------------------------------------------------------
extern "C" void kernel_launch(void* const* d_in, const int* in_sizes, int n_in,
                              void* d_out, int out_size, void* d_ws, size_t ws_size,
                              hipStream_t stream) {
    const float* x    = (const float*)d_in[0];
    const float* wih0 = (const float*)d_in[1];
    const float* whh0 = (const float*)d_in[2];
    const float* bih0 = (const float*)d_in[3];
    const float* bhh0 = (const float*)d_in[4];
    const float* wih1 = (const float*)d_in[5];
    const float* whh1 = (const float*)d_in[6];
    const float* bih1 = (const float*)d_in[7];
    const float* bhh1 = (const float*)d_in[8];
    const float* wout = (const float*)d_in[9];
    const float* bout = (const float*)d_in[10];
    float* out = (float*)d_out;

    const int N = in_sizes[0] / (28 * 28);   // 8192

    // workspace layout
    char* ws = (char*)d_ws;
    size_t off = 0;
    _Float16* wih0f = (_Float16*)(ws + off); off += 512 * 32 * 2;
    _Float16* whh0f = (_Float16*)(ws + off); off += 512 * HID * 2;
    _Float16* wih1f = (_Float16*)(ws + off); off += 512 * HID * 2;
    _Float16* whh1f = (_Float16*)(ws + off); off += 512 * HID * 2;
    float*    bias0 = (float*)   (ws + off); off += 512 * 4;
    float*    bias1 = (float*)   (ws + off); off += 512 * 4;
    _Float16* x16   = (_Float16*)(ws + off); off += (size_t)N * SEQT * 32 * 2;
    _Float16* hs1   = (_Float16*)(ws + off); // N*28*128 f16

    prep_kernel<<<256, 256, 0, stream>>>(x, N * SEQT,
                                         wih0, whh0, bih0, bhh0,
                                         wih1, whh1, bih1, bhh1,
                                         x16, wih0f, whh0f, wih1f, whh1f,
                                         bias0, bias1);

    lstm_layer<0><<<N / 64, 256, 0, stream>>>(x16, wih0f, whh0f, bias0,
                                              hs1, nullptr, nullptr, nullptr);

    lstm_layer<1><<<N / 64, 256, 0, stream>>>(hs1, wih1f, whh1f, bias1,
                                              nullptr, wout, bout, out);
}
